// WaveNet_78529182040384
// MI455X (gfx1250) — matrix operand order repaired and checked by reading, not compiled, not run on hardware
//
#include <hip/hip_runtime.h>

typedef __attribute__((ext_vector_type(16))) _Float16 v16h;
typedef __attribute__((ext_vector_type(8)))  _Float16 v8h;
typedef __attribute__((ext_vector_type(8)))  float    v8f;

// ---------------- problem sizes ----------------
constexpr int kMU1  = 256;     // mu+1 levels
constexpr int kH    = 64;      // hidden
constexpr int kSkip = 256;
constexpr int kNL   = 24;
constexpr int kLF   = 80;
constexpr int kUPK  = 4;
constexpr int kREP  = 75;
constexpr int kB    = 2;
constexpr int kLLEN = 64;
constexpr int kSLEN = 19200;

// ---------------- fused-kernel tiling ----------------
constexpr int kHALO  = 256;            // >= receptive field 252, multiple of 16
constexpr int kTOUT  = 128;            // output cols per block (8 waves x 16)
constexpr int kWIDTH = kHALO + kTOUT;  // 384 = 24 tiles of 16
constexpr int kGUARD = 32;             // zero guard for dilated reads (d<=32)
constexpr int kCOLS  = kGUARD + kWIDTH;// 416
constexpr int kCSTR  = 72;             // sig channel stride (halfs), padded
constexpr int kHSTR  = 264;            // scratch channel stride (halfs), padded

// per-layer LDS weight staging (halfs)
constexpr int kDilL  = 2 * 8 * 2 * 512;   // 16384 halfs = 32KB per layer
constexpr int kResL  = 4 * 2 * 512;       // 4096 halfs  =  8KB per layer
constexpr int kWBUF  = kDilL + kResL;     // 20480 halfs per buffer

// ---------------- workspace layout (bytes) ----------------
constexpr size_t kNDil = (size_t)kNL * kDilL;
constexpr size_t kNSkp = (size_t)kNL * 16 * 2 * 512;
constexpr size_t kNRes = (size_t)(kNL - 1) * kResL;
constexpr size_t kNOut = (size_t)16 * 8 * 512;
constexpr size_t kOffDil  = 0;
constexpr size_t kOffSkp  = kOffDil + kNDil * 2;
constexpr size_t kOffRes  = kOffSkp + kNSkp * 2;
constexpr size_t kOffO1   = kOffRes + kNRes * 2;
constexpr size_t kOffO2   = kOffO1 + kNOut * 2;
constexpr size_t kOffCond = kOffO2 + kNOut * 2;
constexpr size_t kNCond   = (size_t)kB * 256 * kNL * 128;    // floats
constexpr size_t kOffSkb  = kOffCond + kNCond * 4;

// per-layer leftmost tile that must still be computed (halo trimming),
// derived backward from m[last]=16 with m[i-1] = m[i] - ceil(d_i/16)
__device__ __constant__ int kTrim[kNL] =
    {0,0,0,0,0,0, 0,0,0,0,0,2, 3,4,5,6,7,9, 10,11,12,13,14,16};

__device__ __forceinline__ v8f wmma_f16(v16h a, v16h b, v8f c) {
  return __builtin_amdgcn_wmma_f32_16x16x32_f16(false, a, false, b,
                                                (short)0, c, false, false);
}

// CDNA5 async global->LDS copy (ASYNCcnt-tracked, no VGPR data path).
__device__ __forceinline__ void async_copy_b128(unsigned lds_byte_addr,
                                                const void* gaddr) {
  asm volatile("global_load_async_to_lds_b128 %0, %1, off"
               :: "v"(lds_byte_addr),
                  "v"((unsigned long long)(__SIZE_TYPE__)gaddr)
               : "memory");
}
__device__ __forceinline__ void wait_async0() {
  asm volatile("s_wait_asynccnt 0x0" ::: "memory");
}

// A-fragment element mapping used by this code (contiguous k; B-fragments use the same order):
// lane = (half<<4)|row ; element e -> K = half*16+e
__device__ __forceinline__ void frag_mk(int mt, int kt, int lane, int e,
                                        int& m, int& k) {
  m = mt * 16 + (lane & 15);
  int kk = (lane >> 4) * 16 + e;
  k = kt * 32 + kk;
}

// =====================================================================
// Prologue 1: repack all weight matrices (f32) into f16 A-fragments,
// one contiguous 16-half chunk per lane -> b128 loads at use site.
// =====================================================================
__global__ void pack_weights(const float* __restrict__ dilW,
                             const float* __restrict__ skipW,
                             const float* __restrict__ resW,
                             const float* __restrict__ ow1,
                             const float* __restrict__ ow2,
                             _Float16* __restrict__ dilA,
                             _Float16* __restrict__ skipA,
                             _Float16* __restrict__ resA,
                             _Float16* __restrict__ o1A,
                             _Float16* __restrict__ o2A) {
  size_t u = (size_t)blockIdx.x * blockDim.x + threadIdx.x;
  const size_t nD = kNDil, nS = kNSkp, nR = kNRes, nO = kNOut;
  if (u < nD) {                                    // dil_W[i][o][c][sel]
    int e = u & 15, lane = (u >> 4) & 31, kt = (u >> 9) & 1;
    int mt = (u >> 10) & 7, s = (u >> 13) & 1, i = (int)(u >> 14);
    int m, k; frag_mk(mt, kt, lane, e, m, k);
    dilA[u] = (_Float16)dilW[((size_t)(i * 128 + m) * 64 + k) * 2 + s];
    return;
  }
  u -= nD;
  if (u < nS) {                                    // skip_W[i][o][c]
    int e = u & 15, lane = (u >> 4) & 31, kt = (u >> 9) & 1;
    int mt = (u >> 10) & 15, i = (int)(u >> 14);
    int m, k; frag_mk(mt, kt, lane, e, m, k);
    skipA[u] = (_Float16)skipW[(size_t)(i * 256 + m) * 64 + k];
    return;
  }
  u -= nS;
  if (u < nR) {                                    // res_W[i][o][c]
    int e = u & 15, lane = (u >> 4) & 31, kt = (u >> 9) & 1;
    int mt = (u >> 10) & 3, i = (int)(u >> 12);
    int m, k; frag_mk(mt, kt, lane, e, m, k);
    resA[u] = (_Float16)resW[(size_t)(i * 64 + m) * 64 + k];
    return;
  }
  u -= nR;
  if (u < 2 * nO) {                                // out_W1 / out_W2 [256][256]
    const float* src = (u < nO) ? ow1 : ow2;
    _Float16* dst = (u < nO) ? o1A : o2A;
    size_t v = (u < nO) ? u : (u - nO);
    int e = v & 15, lane = (v >> 4) & 31, kt = (v >> 9) & 7, mt = (int)(v >> 12);
    int m, k; frag_mk(mt, kt, lane, e, m, k);
    dst[v] = (_Float16)src[(size_t)m * 256 + k];
  }
}

// =====================================================================
// Prologue 2: conditional-feature table.  cond has only 256 distinct time
// frames per batch (upsample k=4 then repeat 75).  Fold dil_b in.
// condfold[b][frame][layer][128]
// =====================================================================
__global__ void build_cond(const float* __restrict__ lf,
                           const float* __restrict__ upW,
                           const float* __restrict__ upB,
                           const float* __restrict__ condW,
                           const float* __restrict__ condB,
                           const float* __restrict__ dilB,
                           float* __restrict__ condfold) {
  __shared__ float up[kLF];
  int f = blockIdx.x, b = blockIdx.y;
  int tp = f >> 2, kq = f & 3;
  for (int o = threadIdx.x; o < kLF; o += blockDim.x) {
    float s = upB[o];
    for (int c = 0; c < kLF; ++c)
      s += lf[((size_t)b * kLLEN + tp) * kLF + c] * upW[(c * kLF + o) * kUPK + kq];
    up[o] = s;
  }
  __syncthreads();
  for (int oc = threadIdx.x; oc < kNL * 2 * kH; oc += blockDim.x) {
    float s = condB[oc];
    for (int o = 0; o < kLF; ++o) s += condW[(size_t)oc * kLF + o] * up[o];
    int il = oc >> 7, m = oc & 127;
    s += dilB[il * 128 + m];
    condfold[(((size_t)b * 256 + f) * kNL + il) * 128 + m] = s;
  }
}

// Prologue 3: skip biases are layer-summed constants per channel.
__global__ void build_skipb(const float* __restrict__ skipB,
                            float* __restrict__ skipbsum) {
  int c = threadIdx.x;
  float s = 0.f;
  for (int i = 0; i < kNL; ++i) s += skipB[i * kSkip + c];
  skipbsum[c] = s;
}

// =====================================================================
// Fused WaveNet: embed -> 24 residual layers -> head -> log_softmax.
// 256 threads = 8 wave32; each wave owns one 16-col output tile whose
// 256-channel skip accumulator lives in 16 WMMA C-fragments (VGPRs).
// Gate+res weight fragments are async-staged into LDS, double-buffered.
// =====================================================================
__global__ __launch_bounds__(256, 1)
void wavenet_fused(const int* __restrict__ gold,
                   const float* __restrict__ embW, const float* __restrict__ embB,
                   const _Float16* __restrict__ dilA,
                   const _Float16* __restrict__ skipA,
                   const _Float16* __restrict__ resA,
                   const _Float16* __restrict__ out1A,
                   const _Float16* __restrict__ out2A,
                   const float* __restrict__ condfold,
                   const float* __restrict__ skipbsum,
                   const float* __restrict__ resb,
                   float* __restrict__ out) {
  extern __shared__ _Float16 smem[];
  _Float16* sig  = smem;                            // [2][kCOLS][kCSTR] f16
  _Float16* sc   = smem + 2 * kCOLS * kCSTR;        // [8 waves][16 cols][kHSTR]
  _Float16* wbuf = sc + 8 * 16 * kHSTR;             // [2][kWBUF] staged weights
  const int b    = blockIdx.y;
  const int t0   = blockIdx.x * kTOUT;              // first output column
  const int tid  = threadIdx.x;
  const int wv   = tid >> 5;
  const int lane = tid & 31;
  const int lo   = lane & 15;                       // column within tile
  const int hi   = lane >> 4;                       // lane half

  // ---- kick off async staging of layer-0 gate/res fragments ----
  {
    const char* gsrc = (const char*)dilA;           // layer 0 chunk
    unsigned ldst = (unsigned)(__SIZE_TYPE__)wbuf;
    for (int c = tid; c < kDilL / 8; c += 256)      // 16B chunks
      async_copy_b128(ldst + c * 16, gsrc + c * 16);
    const char* grsrc = (const char*)resA;
    unsigned lrdst = (unsigned)(__SIZE_TYPE__)(wbuf + kDilL);
    for (int c = tid; c < kResL / 8; c += 256)
      async_copy_b128(lrdst + c * 16, grsrc + c * 16);
  }

  // ---- embed mu-law one-hot (gather) into sig buffer 0; zero guards ----
  for (int task = tid; task < kCOLS * 8; task += 256) {
    int col = task >> 3, ch0 = (task & 7) * 8;
    int t = t0 - kHALO + (col - kGUARD);
    v8h v;
#pragma unroll
    for (int j = 0; j < 8; ++j) v[j] = (_Float16)0.f;
    if (col >= kGUARD && t >= 0) {
      int idx = gold[(size_t)b * kSLEN + t];
#pragma unroll
      for (int j = 0; j < 8; ++j)
        v[j] = (_Float16)(embW[(size_t)(ch0 + j) * kMU1 + idx] + embB[ch0 + j]);
    }
    *(v8h*)&sig[(size_t)col * kCSTR + ch0] = v;
    if (col < kGUARD)                               // zero guard of buffer 1 too
      *(v8h*)&sig[(size_t)(kCOLS + col) * kCSTR + ch0] = v;
  }
  wait_async0();
  __syncthreads();

  // ---- persistent skip accumulators (C-fragment layout) ----
  v8f skip_acc[16];
#pragma unroll
  for (int mt = 0; mt < 16; ++mt)
#pragma unroll
    for (int r = 0; r < 8; ++r)
      skip_acc[mt][r] = skipbsum[mt * 16 + hi * 8 + r];

  _Float16* zbuf = sc + ((size_t)(wv * 16 + lo)) * kHSTR;  // this lane's column
  int cur = 0, curw = 0;

  for (int i = 0; i < kNL; ++i) {
    const int d = 1 << (i % 6);
    const int mtrim = kTrim[i];
    const _Float16* wd = wbuf + (size_t)curw * kWBUF;        // staged dil frags
    const _Float16* wr = wd + kDilL;                         // staged res frags
    const _Float16* sA = skipA + (size_t)i * (16 * 2 * 512); // skip from L2

    // async-prefetch next layer's gate/res fragments into the other buffer
    if (i + 1 < kNL) {
      const char* gsrc = (const char*)(dilA + (size_t)(i + 1) * kDilL);
      unsigned ldst = (unsigned)(__SIZE_TYPE__)(wbuf + (size_t)(curw ^ 1) * kWBUF);
      for (int c = tid; c < kDilL / 8; c += 256)
        async_copy_b128(ldst + c * 16, gsrc + c * 16);
      if (i + 1 < kNL - 1) {
        const char* grsrc = (const char*)(resA + (size_t)(i + 1) * kResL);
        unsigned lrdst = ldst + kDilL * 2;
        for (int c = tid; c < kResL / 8; c += 256)
          async_copy_b128(lrdst + c * 16, grsrc + c * 16);
      }
      __builtin_prefetch(skipA + (size_t)(i + 1) * (16 * 2 * 512), 0, 1);
    }

    const _Float16* srcb = sig + (size_t)cur * kCOLS * kCSTR;
    _Float16*       dstb = sig + (size_t)(cur ^ 1) * kCOLS * kCSTR;

    for (int j = 0; j < 3; ++j) {                   // tiles wv, wv+8, wv+16
      const int tile = wv + j * 8;
      const bool isOut = (j == 2);
      if (!isOut && tile < mtrim) continue;         // halo trimming
      const int colL = tile * 16 + lo;
      const int tg = t0 - kHALO + colL;             // global time of this column

      // B-fragments: current sig and dilated (causal) sig
      const _Float16* pc = srcb + (size_t)(kGUARD + colL) * kCSTR;
      const _Float16* pp = srcb + (size_t)(kGUARD + colL - d) * kCSTR;
      v16h bs0 = *(const v16h*)(pc + hi * 16);
      v16h bs1 = *(const v16h*)(pc + 32 + hi * 16);
      v16h bp0 = *(const v16h*)(pp + hi * 16);
      v16h bp1 = *(const v16h*)(pp + 32 + hi * 16);

      const int f = (tg >= 0) ? (tg / kREP) : 0;
      const float* cf = condfold + (((size_t)(b * 256 + f)) * kNL + i) * 128;

      // ---- gated activation: a = W0@prev + W1@sig + (cond+bias) ----
#pragma unroll
      for (int mt = 0; mt < 4; ++mt) {
        v8f clo, chi;
#pragma unroll
        for (int r = 0; r < 8; ++r) {
          clo[r] = cf[mt * 16 + hi * 8 + r];
          chi[r] = cf[64 + mt * 16 + hi * 8 + r];
        }
        v16h a;
        a = *(const v16h*)(wd + ((size_t)((0 * 8 + mt) * 2 + 0)) * 512 + lane * 16);
        clo = wmma_f16(a, bp0, clo);
        a = *(const v16h*)(wd + ((size_t)((0 * 8 + mt) * 2 + 1)) * 512 + lane * 16);
        clo = wmma_f16(a, bp1, clo);
        a = *(const v16h*)(wd + ((size_t)((1 * 8 + mt) * 2 + 0)) * 512 + lane * 16);
        clo = wmma_f16(a, bs0, clo);
        a = *(const v16h*)(wd + ((size_t)((1 * 8 + mt) * 2 + 1)) * 512 + lane * 16);
        clo = wmma_f16(a, bs1, clo);
        a = *(const v16h*)(wd + ((size_t)((0 * 8 + mt + 4) * 2 + 0)) * 512 + lane * 16);
        chi = wmma_f16(a, bp0, chi);
        a = *(const v16h*)(wd + ((size_t)((0 * 8 + mt + 4) * 2 + 1)) * 512 + lane * 16);
        chi = wmma_f16(a, bp1, chi);
        a = *(const v16h*)(wd + ((size_t)((1 * 8 + mt + 4) * 2 + 0)) * 512 + lane * 16);
        chi = wmma_f16(a, bs0, chi);
        a = *(const v16h*)(wd + ((size_t)((1 * 8 + mt + 4) * 2 + 1)) * 512 + lane * 16);
        chi = wmma_f16(a, bs1, chi);
        v8h z;
#pragma unroll
        for (int r = 0; r < 8; ++r) {  // z = tanh(lo)*sigmoid(hi), fast rcp
          float ex = __expf(2.f * clo[r]);
          float th = 1.f - 2.f * __builtin_amdgcn_rcpf(ex + 1.f);
          float sg = __builtin_amdgcn_rcpf(1.f + __expf(-chi[r]));
          z[r] = (_Float16)(th * sg);
        }
        *(v8h*)&zbuf[mt * 16 + hi * 8] = z;         // wave-local LDS transpose
      }

      v16h z0 = *(const v16h*)(zbuf + hi * 16);     // z as B-fragments
      v16h z1 = *(const v16h*)(zbuf + 32 + hi * 16);

      if (isOut) {                                  // skip accumulation (regs)
#pragma unroll
        for (int mt = 0; mt < 16; ++mt) {
          v16h a0 = *(const v16h*)(sA + ((size_t)(mt * 2 + 0)) * 512 + lane * 16);
          skip_acc[mt] = wmma_f16(a0, z0, skip_acc[mt]);
          v16h a1 = *(const v16h*)(sA + ((size_t)(mt * 2 + 1)) * 512 + lane * 16);
          skip_acc[mt] = wmma_f16(a1, z1, skip_acc[mt]);
        }
      }
      if (i < kNL - 1) {                            // residual -> next buffer
#pragma unroll
        for (int mt = 0; mt < 4; ++mt) {
          v8f rc;
#pragma unroll
          for (int r = 0; r < 8; ++r) rc[r] = resb[i * 64 + mt * 16 + hi * 8 + r];
          v16h a0 = *(const v16h*)(wr + ((size_t)(mt * 2 + 0)) * 512 + lane * 16);
          rc = wmma_f16(a0, z0, rc);
          v16h a1 = *(const v16h*)(wr + ((size_t)(mt * 2 + 1)) * 512 + lane * 16);
          rc = wmma_f16(a1, z1, rc);
          v8h old = *(const v8h*)(pc + mt * 16 + hi * 8);
          v8h nw;
#pragma unroll
          for (int r = 0; r < 8; ++r)               // keep t<0 exactly zero
            nw[r] = (tg >= 0) ? (_Float16)((float)old[r] + rc[r]) : (_Float16)0.f;
          *(v8h*)(dstb + (size_t)(kGUARD + colL) * kCSTR + mt * 16 + hi * 8) = nw;
        }
      }
    }
    wait_async0();                                  // next-layer weights landed
    __syncthreads();
    cur ^= 1;
    curw ^= 1;
  }

  // ---- head: relu -> out_W1 -> relu -> out_W2 -> log_softmax ----
#pragma unroll
  for (int mt = 0; mt < 16; ++mt) {                 // stage relu(skip) as f16
    v8h v;
#pragma unroll
    for (int r = 0; r < 8; ++r) {
      float x = skip_acc[mt][r];
      v[r] = (_Float16)(x > 0.f ? x : 0.f);
    }
    *(v8h*)&zbuf[mt * 16 + hi * 8] = v;
  }
  v16h hf[8];
#pragma unroll
  for (int kt = 0; kt < 8; ++kt)
    hf[kt] = *(const v16h*)(zbuf + kt * 32 + hi * 16);
#pragma unroll
  for (int mt = 0; mt < 16; ++mt) {                 // h1 = relu(W1 @ h)
    v8f acc;
#pragma unroll
    for (int r = 0; r < 8; ++r) acc[r] = 0.f;
#pragma unroll
    for (int kt = 0; kt < 8; ++kt) {
      v16h a = *(const v16h*)(out1A + ((size_t)(mt * 8 + kt)) * 512 + lane * 16);
      acc = wmma_f16(a, hf[kt], acc);
    }
    v8h v;
#pragma unroll
    for (int r = 0; r < 8; ++r) {
      float x = acc[r];
      v[r] = (_Float16)(x > 0.f ? x : 0.f);
    }
    *(v8h*)&zbuf[mt * 16 + hi * 8] = v;
  }
#pragma unroll
  for (int kt = 0; kt < 8; ++kt)
    hf[kt] = *(const v16h*)(zbuf + kt * 32 + hi * 16);
  v8f lg[16];
#pragma unroll
  for (int mt = 0; mt < 16; ++mt) {                 // logits = W2 @ h1
    v8f acc;
#pragma unroll
    for (int r = 0; r < 8; ++r) acc[r] = 0.f;
#pragma unroll
    for (int kt = 0; kt < 8; ++kt) {
      v16h a = *(const v16h*)(out2A + ((size_t)(mt * 8 + kt)) * 512 + lane * 16);
      acc = wmma_f16(a, hf[kt], acc);
    }
    lg[mt] = acc;
  }

  // log_softmax over the 256 channels of this lane's column (split between
  // lane and lane^16 -> one wave32 xor-shuffle finishes the reduction).
  float mx = -1e30f;
#pragma unroll
  for (int mt = 0; mt < 16; ++mt)
#pragma unroll
    for (int r = 0; r < 8; ++r) mx = fmaxf(mx, lg[mt][r]);
  mx = fmaxf(mx, __shfl_xor(mx, 16, 32));
  float se = 0.f;
#pragma unroll
  for (int mt = 0; mt < 16; ++mt)
#pragma unroll
    for (int r = 0; r < 8; ++r) se += __expf(lg[mt][r] - mx);
  se += __shfl_xor(se, 16, 32);
  const float lse = __logf(se) + mx;

  const int tcol = t0 + wv * 16 + lo;               // this lane's output column
#pragma unroll
  for (int mt = 0; mt < 16; ++mt)
#pragma unroll
    for (int r = 0; r < 8; ++r) {
      int m = mt * 16 + hi * 8 + r;
      out[((size_t)b * kMU1 + m) * kSLEN + tcol] = lg[mt][r] - lse;
    }
}

// =====================================================================
extern "C" void kernel_launch(void* const* d_in, const int* in_sizes, int n_in,
                              void* d_out, int out_size, void* d_ws,
                              size_t ws_size, hipStream_t stream) {
  (void)in_sizes; (void)n_in; (void)out_size; (void)ws_size;
  const float* lf    = (const float*)d_in[0];
  const int*   gold  = (const int*)d_in[1];
  const float* embW  = (const float*)d_in[2];
  const float* embB  = (const float*)d_in[3];
  const float* upW   = (const float*)d_in[4];
  const float* upB   = (const float*)d_in[5];
  const float* condW = (const float*)d_in[6];
  const float* condB = (const float*)d_in[7];
  const float* dilW  = (const float*)d_in[8];
  const float* dilB  = (const float*)d_in[9];
  const float* skipW = (const float*)d_in[10];
  const float* skipB = (const float*)d_in[11];
  const float* resW  = (const float*)d_in[12];
  const float* resB  = (const float*)d_in[13];
  const float* ow1   = (const float*)d_in[14];
  const float* ow2   = (const float*)d_in[15];
  float* out = (float*)d_out;
  char*  ws  = (char*)d_ws;

  _Float16* dilA  = (_Float16*)(ws + kOffDil);
  _Float16* skipA = (_Float16*)(ws + kOffSkp);
  _Float16* resA  = (_Float16*)(ws + kOffRes);
  _Float16* o1A   = (_Float16*)(ws + kOffO1);
  _Float16* o2A   = (_Float16*)(ws + kOffO2);
  float* condfold = (float*)(ws + kOffCond);
  float* skipbsum = (float*)(ws + kOffSkb);

  const size_t nPack = kNDil + kNSkp + kNRes + 2 * kNOut;
  pack_weights<<<dim3((unsigned)((nPack + 255) / 256)), 256, 0, stream>>>(
      dilW, skipW, resW, ow1, ow2, dilA, skipA, resA, o1A, o2A);
  build_cond<<<dim3(256, kB), 128, 0, stream>>>(lf, upW, upB, condW, condB,
                                                dilB, condfold);
  build_skipb<<<1, 256, 0, stream>>>(skipB, skipbsum);

  const size_t shmem = ((size_t)2 * kCOLS * kCSTR + (size_t)8 * 16 * kHSTR +
                        (size_t)2 * kWBUF) * sizeof(_Float16);  // 269312 B
  wavenet_fused<<<dim3(kSLEN / kTOUT, kB), 256, shmem, stream>>>(
      gold, embW, embB, dilA, skipA, resA, o1A, o2A, condfold, skipbsum, resB,
      out);
}
